// Quadratic_operation_86397562126642
// MI455X (gfx1250) — matrix operand order repaired and checked by reading, not compiled, not run on hardware
//
#include <hip/hip_runtime.h>
#include <hip/hip_bf16.h>

// ---------------------------------------------------------------------------
// out[16384,512] = quad_expand(x[16384,128]) @ W[8384,512] + bias[512]
//
// prep_pairs : byte-offset (i*4, j*4) table for the 8384 features -> d_ws
// prep_wfrag : W fp32 -> bf16, pre-swizzled into WMMA B-fragment layout -> d_ws
// quad_gemm  : fused feature-gen (LDS, double-buffered) + bf16 WMMA GEMM
//              wave tile 4M x 2N (64 rows x 32 cols) for 4x B-fragment reuse
// ---------------------------------------------------------------------------

typedef __bf16 v16bf __attribute__((ext_vector_type(16)));
typedef float  v8f   __attribute__((ext_vector_type(8)));

#define BATCH   16384
#define N_IN    128
#define N_FEAT  8384              // 128 + 128*129/2
#define N_OUT   512
#define N_KSTEP (N_FEAT / 32)     // 262 WMMA k-steps
#define N_NTILE (N_OUT / 16)      // 32

#define BM      128               // block M tile
#define BN      256               // block N tile (grid.y = 2)
#define BK      64                // K per iteration (2 WMMA k-steps)
#define NITER   (N_FEAT / BK)     // 131
#define NTHREADS 512              // 16 waves

#define XS_STRIDE 132             // padded fp32 row stride (129 used)
#define XS_BYTES  (BM * XS_STRIDE * 4)          // 67584
#define FRAGW     (8 * 2 * 32 * 8)              // A frag words per buffer
#define SMEM_BYTES (XS_BYTES + 2 * FRAGW * 4)   // 67584 + 32768 = 100352

// workspace layout
#define WS_PAIRS_OFF 0
#define WS_WFRAG_OFF 65536
#define WFRAG_WORDS  ((size_t)N_KSTEP * N_NTILE * 32 * 8)   // 2,146,304 (8.6 MB)

__device__ __forceinline__ unsigned f2bf_bits(float f) {
    unsigned u = __float_as_uint(f);
    u += 0x7FFFu + ((u >> 16) & 1u);            // round-to-nearest-even
    return u >> 16;
}
__device__ __forceinline__ unsigned pack2(float lo, float hi) {
    return f2bf_bits(lo) | (f2bf_bits(hi) << 16);
}

// ---- prep: feature pair table (pre-scaled byte offsets) -------------------
__global__ __launch_bounds__(256)
void prep_pairs(unsigned* __restrict__ pairs) {
    int tid = threadIdx.x;
    for (int k = tid; k < N_IN; k += 256)       // linear term: x[k] * 1.0
        pairs[k] = (unsigned)(k * 4) | ((128u * 4) << 16);
    if (tid < N_IN) {
        int i = tid;
        int off = N_IN + i * N_IN - (i * (i - 1)) / 2;
        for (int j = i; j < N_IN; ++j)
            pairs[off + (j - i)] = (unsigned)(i * 4) | ((unsigned)(j * 4) << 16);
    }
}

// ---- prep: W -> bf16 B-fragment layout [kstep][ntile][lane][v] -----------
__global__ __launch_bounds__(256)
void prep_wfrag(const float* __restrict__ w, unsigned* __restrict__ wfrag) {
    int ks = blockIdx.x;                        // 0..261
    for (int idx = threadIdx.x; idx < N_NTILE * 32 * 8; idx += 256) {
        int ntile = idx >> 8;
        int lane  = (idx >> 3) & 31;
        int v     = idx & 7;
        int col = ntile * 16 + (lane & 15);
        int k   = ks * 32 + 2 * (v & 3) + (v >> 2) * 16 + (lane >> 4) * 8;   // B: lanes16-31 K+8
        unsigned word = pack2(w[(size_t)k * N_OUT + col],
                              w[(size_t)(k + 1) * N_OUT + col]);
        wfrag[(size_t)ks * (N_NTILE * 32 * 8) + idx] = word;
    }
}

// ---- main fused GEMM ------------------------------------------------------
__global__ __launch_bounds__(NTHREADS)
void quad_gemm_kernel(const float* __restrict__ x,
                      const float* __restrict__ bias,
                      const unsigned* __restrict__ pairs,
                      const unsigned* __restrict__ wfrag,
                      float* __restrict__ out) {
    extern __shared__ char smem[];
    float*    xs  = (float*)smem;                       // [BM][XS_STRIDE]
    unsigned* afr = (unsigned*)(smem + XS_BYTES);       // 2 x [8][2][32][8]

    const int tid  = threadIdx.x;
    const int lane = tid & 31;
    const int wid  = tid >> 5;                          // 16 waves
    const int hi   = lane >> 4;
    const int l15  = lane & 15;

    const int rowBase = blockIdx.x * BM;
    const int nBase   = blockIdx.y * (BN / 16);         // first ntile of block

    // ---- load x tile (coalesced float4), append ones column
    for (int idx = tid; idx < BM * (N_IN / 4); idx += NTHREADS) {
        int r  = idx >> 5;                              // N_IN/4 == 32
        int c4 = idx & 31;
        float4 v = ((const float4*)(x + (size_t)(rowBase + r) * N_IN))[c4];
        float* dst = xs + r * XS_STRIDE + c4 * 4;
        dst[0] = v.x; dst[1] = v.y; dst[2] = v.z; dst[3] = v.w;
    }
    if (tid < BM) xs[tid * XS_STRIDE + 128] = 1.0f;

    // ---- wave tile: 4 M-subtiles x 2 N-subtiles (64 rows x 32 cols)
    const int mg = wid & 1;                             // mtiles 4*mg..4*mg+3
    const int ng = wid >> 1;                            // ntiles nBase+2*ng..+1

    v8f acc[4][2];
    #pragma unroll
    for (int m = 0; m < 4; ++m)
        #pragma unroll
        for (int n = 0; n < 2; ++n)
            #pragma unroll
            for (int e = 0; e < 8; ++e) acc[m][n][e] = 0.0f;

    // ---- A generation: thread owns (mtile = wid&7, kstep = wid>>3), 8 words
    const int genM  = wid & 7;
    const int genKs = wid >> 3;
    const char* xrow = (const char*)(xs + (genM * 16 + l15) * XS_STRIDE);
    const int genBase = ((genM * 2 + genKs) * 32 + lane) * 8;

    auto genA = [&](int kb, int buf) {
        unsigned* dst = afr + buf * FRAGW + genBase;
        #pragma unroll
        for (int v = 0; v < 8; ++v) {
            int klocal = ((v < 4) ? (2 * v) : (16 + 2 * (v - 4))) + hi * 8;
            int k0 = kb + genKs * 32 + klocal;          // always even
            uint2 pr = *(const uint2*)(pairs + k0);
            float f0 = *(const float*)(xrow + (pr.x & 0xFFFFu)) *
                       *(const float*)(xrow + (pr.x >> 16));
            float f1 = *(const float*)(xrow + (pr.y & 0xFFFFu)) *
                       *(const float*)(xrow + (pr.y >> 16));
            dst[v] = pack2(f0, f1);
        }
    };

    __syncthreads();           // xs ready
    genA(0, 0);
    __syncthreads();           // buffer 0 ready

    const v16bf* wfragV = (const v16bf*)wfrag;

    for (int it = 0; it < NITER; ++it) {
        const int kb = it * BK;
        if (it + 1 < NITER) {
            genA(kb + BK, (it + 1) & 1);                // overlap with WMMA
            __builtin_prefetch(wfrag + ((size_t)((kb >> 5) + 4) * N_NTILE +
                                        nBase + 2 * ng) * 32 * 8, 0, 3);
        }
        const v16bf* abuf = (const v16bf*)(afr + (it & 1) * FRAGW);
        #pragma unroll
        for (int ks = 0; ks < 2; ++ks) {
            const int ksg = (kb >> 5) + ks;
            // hoist all fragment loads for this k-step, then the WMMA chain
            v16bf a[4];
            #pragma unroll
            for (int m = 0; m < 4; ++m)
                a[m] = abuf[((4 * mg + m) * 2 + ks) * 32 + lane];
            v16bf b[2];
            #pragma unroll
            for (int n = 0; n < 2; ++n)
                b[n] = wfragV[((size_t)ksg * N_NTILE + nBase + 2 * ng + n) * 32
                              + lane];
            #pragma unroll
            for (int n = 0; n < 2; ++n)
                #pragma unroll
                for (int m = 0; m < 4; ++m)
                    acc[m][n] = __builtin_amdgcn_wmma_f32_16x16x32_bf16(
                        false, a[m], false, b[n], (short)0, acc[m][n],
                        false, false);
        }
        __syncthreads();
    }

    // ---- epilogue: bias + store (C: VGPR r -> M = r + 8*hi, N = lane&15)
    #pragma unroll
    for (int n = 0; n < 2; ++n) {
        int col = (nBase + 2 * ng + n) * 16 + l15;
        float bz = bias[col];
        #pragma unroll
        for (int m = 0; m < 4; ++m) {
            int rbase = rowBase + (4 * mg + m) * 16 + hi * 8;
            #pragma unroll
            for (int r = 0; r < 8; ++r)
                out[(size_t)(rbase + r) * N_OUT + col] = acc[m][n][r] + bz;
        }
    }
}

extern "C" void kernel_launch(void* const* d_in, const int* in_sizes, int n_in,
                              void* d_out, int out_size, void* d_ws, size_t ws_size,
                              hipStream_t stream) {
    const float* x    = (const float*)d_in[0];
    const float* wgt  = (const float*)d_in[1];
    const float* bias = (const float*)d_in[2];
    float* out = (float*)d_out;

    unsigned* pairs = (unsigned*)((char*)d_ws + WS_PAIRS_OFF);
    unsigned* wfrag = (unsigned*)((char*)d_ws + WS_WFRAG_OFF);

    (void)in_sizes; (void)n_in; (void)out_size; (void)ws_size;

    prep_pairs<<<dim3(1), dim3(256), 0, stream>>>(pairs);
    prep_wfrag<<<dim3(N_KSTEP), dim3(256), 0, stream>>>(wgt, wfrag);

    hipFuncSetAttribute(reinterpret_cast<const void*>(quad_gemm_kernel),
                        hipFuncAttributeMaxDynamicSharedMemorySize, SMEM_BYTES);
    dim3 grid(BATCH / BM, N_OUT / BN);                  // 128 x 2
    quad_gemm_kernel<<<grid, dim3(NTHREADS), SMEM_BYTES, stream>>>(
        x, bias, pairs, wfrag, out);
}
